// ImprovedProteinPocketEncoder_34084860461599
// MI455X (gfx1250) — matrix operand order repaired and checked by reading, not compiled, not run on hardware
//
#include <hip/hip_runtime.h>
#include <hip/hip_fp16.h>
#include <stdint.h>

// ---------------------------------------------------------------------------
// ImprovedProteinPocketEncoder for MI455X (gfx1250, wave32, WMMA).
// Heavy per-edge MLPs run as fused f16 WMMA GEMMs (f32 accum); segment sums
// via global_atomic_add_f32; small M=1 / K-tiny ops on VALU.
// Round 2: 2 M-tiles per wave (B-fragment reuse x2, dual independent WMMA
// chains for latency hiding), fast rcp-based SiLU/tanh epilogues.
// ---------------------------------------------------------------------------

typedef __attribute__((ext_vector_type(16))) _Float16 v16h;
typedef __attribute__((ext_vector_type(8)))  float    v8f;
typedef _Float16 half_t;

static constexpr int Nn  = 50000;
static constexpr int Ee  = 800000;

__device__ __forceinline__ float fast_rcp(float x) { return __builtin_amdgcn_rcpf(x); }
__device__ __forceinline__ float siluf(float x) { return x * fast_rcp(1.0f + __expf(-x)); }
// tanh(x) = 1 - 2/(exp(2x)+1); exp overflow -> +1, exp->0 -> -1 (correct saturation)
__device__ __forceinline__ float tanh_fast(float x) {
  float t = __expf(2.0f * x);
  return 1.0f - 2.0f * fast_rcp(t + 1.0f);
}

__device__ __forceinline__ v8f wmma32(v16h a, v16h b, v8f c) {
  return __builtin_amdgcn_wmma_f32_16x16x32_f16(false, a, false, b, (short)0, c, false, false);
}

union FragU { v16h v; uint4 q[2]; };

// A fragment (16x32 f16): lane (hi,ml) -> row ml, K halves {c0..c0+7, c0+16..c0+23},
// c0 = kt*32 + 8*hi. Two contiguous 16B chunks.
__device__ __forceinline__ v16h load_afrag(const half_t* rowbase, int kt, int hi) {
  const half_t* p = rowbase + kt * 32 + 8 * hi;
  FragU u;
  u.q[0] = *(const uint4*)(p);
  u.q[1] = *(const uint4*)(p + 16);
  return u.v;
}
// B fragment from packed weights: packed[((kt*NT+nt)*32+lane)*16 + e] = W[kt*32+8*hi+(e&7)+16*(e>>3)][nt*16+nl]
__device__ __forceinline__ v16h load_bfrag(const half_t* Wp, int kt, int nt, int NT, int lane) {
  const half_t* p = Wp + ((size_t)((kt * NT + nt) * 32 + lane)) * 16;
  FragU u;
  u.q[0] = *(const uint4*)(p);
  u.q[1] = *(const uint4*)(p + 8);
  return u.v;
}

// ---------------------------------------------------------------------------
// Weight repack: fp32 row-major (K x N2) -> f16 WMMA B-fragment order.
// ---------------------------------------------------------------------------
__global__ void pack_kernel(const float* __restrict__ W, half_t* __restrict__ out, int K, int N2) {
  int i = blockIdx.x * blockDim.x + threadIdx.x;
  if (i >= K * N2) return;
  int NT = N2 >> 4;
  int e = i & 15;
  int lane = (i >> 4) & 31;
  int ktnt = i >> 9;
  int nt = ktnt % NT, kt = ktnt / NT;
  int hi = lane >> 4, nl = lane & 15;
  int kk = kt * 32 + hi * 8 + (e & 7) + ((e >> 3) << 4);
  int nn = nt * 16 + nl;
  out[i] = (half_t)W[(size_t)kk * N2 + nn];
}

// ---------------------------------------------------------------------------
// Fused EGNN edge kernel. 4 waves/block, 32 edges per wave (2 M-tiles):
//   m1 = silu(h[row]@W1a + h[col]@W1b + rn*w1c + b1)
//   m  = m1@W2 + b2 ; atomic agg[row]+=m
//   c1 = silu(m@Wc1 + cb1); cd = tanh(dot(c1,wc2)+cb2)*0.1
//   atomic pos_delta[row] += cd*radial/rn
// ---------------------------------------------------------------------------
__global__ __launch_bounds__(128) void edge_kernel(
    const half_t* __restrict__ hf16, const float* __restrict__ pos,
    const int* __restrict__ rowI, const int* __restrict__ colI,
    const half_t* __restrict__ W1a, const half_t* __restrict__ W1b,
    const float* __restrict__ w1c, const float* __restrict__ b1,
    const half_t* __restrict__ W2, const float* __restrict__ b2,
    const half_t* __restrict__ Wc1, const float* __restrict__ cb1,
    const float* __restrict__ wc2, const float* __restrict__ cb2,
    float* __restrict__ agg, float* __restrict__ posDelta, int Etot) {
  __shared__ half_t stag[4][32 * 128];   // 32KB: per-wave 32x128 f16 staging
  __shared__ float rns[4][32];

  int wave = threadIdx.x >> 5, lane = threadIdx.x & 31;
  int hi = lane >> 4, ml = lane & 15;
  int e0 = blockIdx.x * 128 + wave * 32;   // Etot is a multiple of 128
  if (e0 >= Etot) return;

  // radial for this lane's edge (e0+lane); also used in coord epilogue
  int eR = e0 + lane;
  int rR = rowI[eR], cR = colI[eR];
  float dx = pos[rR * 3 + 0] - pos[cR * 3 + 0];
  float dy = pos[rR * 3 + 1] - pos[cR * 3 + 1];
  float dz = pos[rR * 3 + 2] - pos[cR * 3 + 2];
  float rn = fmaxf(sqrtf(dx * dx + dy * dy + dz * dz), 1e-8f);
  rns[wave][lane] = rn;

  // A-row indices for the two M-tiles
  int rA[2], cA[2];
#pragma unroll
  for (int t = 0; t < 2; t++) {
    rA[t] = rowI[e0 + t * 16 + ml];
    cA[t] = colI[e0 + t * 16 + ml];
  }
  int idx8[2][8];
#pragma unroll
  for (int t = 0; t < 2; t++)
#pragma unroll
    for (int r = 0; r < 8; r++) idx8[t][r] = rowI[e0 + t * 16 + r + 8 * hi];
  float rnv[2][8];
#pragma unroll
  for (int t = 0; t < 2; t++)
#pragma unroll
    for (int r = 0; r < 8; r++) rnv[t][r] = rns[wave][t * 16 + r + 8 * hi];  // per-wave LDS, in-order

  v16h ar[2][4], ac[2][4];
#pragma unroll
  for (int t = 0; t < 2; t++) {
    const half_t* rb = hf16 + (size_t)rA[t] * 128;
    const half_t* cb = hf16 + (size_t)cA[t] * 128;
#pragma unroll
    for (int kt = 0; kt < 4; kt++) { ar[t][kt] = load_afrag(rb, kt, hi); ac[t][kt] = load_afrag(cb, kt, hi); }
  }

  // ---- GEMM1: hidden of edge MLP ----
#pragma unroll 2
  for (int nt = 0; nt < 8; nt++) {
    v16h bfr[8];
#pragma unroll
    for (int kt = 0; kt < 4; kt++) bfr[kt] = load_bfrag(W1a, kt, nt, 8, lane);
#pragma unroll
    for (int kt = 0; kt < 4; kt++) bfr[4 + kt] = load_bfrag(W1b, kt, nt, 8, lane);
    v8f acc0 = {}, acc1 = {};
#pragma unroll
    for (int kt = 0; kt < 4; kt++) {
      acc0 = wmma32(ar[0][kt], bfr[kt], acc0);
      acc1 = wmma32(ar[1][kt], bfr[kt], acc1);
    }
#pragma unroll
    for (int kt = 0; kt < 4; kt++) {
      acc0 = wmma32(ac[0][kt], bfr[4 + kt], acc0);
      acc1 = wmma32(ac[1][kt], bfr[4 + kt], acc1);
    }
    int n = nt * 16 + ml;
    float bb = b1[n], wc = w1c[n];
#pragma unroll
    for (int r = 0; r < 8; r++) {
      stag[wave][(r + 8 * hi) * 128 + n]      = (half_t)siluf(acc0[r] + bb + rnv[0][r] * wc);
      stag[wave][(16 + r + 8 * hi) * 128 + n] = (half_t)siluf(acc1[r] + bb + rnv[1][r] * wc);
    }
  }

  // ---- GEMM2: m = silu_hidden @ W2 + b2 ----
  v16h a2[2][4];
#pragma unroll
  for (int t = 0; t < 2; t++)
#pragma unroll
    for (int kt = 0; kt < 4; kt++) a2[t][kt] = load_afrag(&stag[wave][(t * 16 + ml) * 128], kt, hi);
#pragma unroll 2
  for (int nt = 0; nt < 8; nt++) {
    v16h bfr[4];
#pragma unroll
    for (int kt = 0; kt < 4; kt++) bfr[kt] = load_bfrag(W2, kt, nt, 8, lane);
    v8f acc0 = {}, acc1 = {};
#pragma unroll
    for (int kt = 0; kt < 4; kt++) {
      acc0 = wmma32(a2[0][kt], bfr[kt], acc0);
      acc1 = wmma32(a2[1][kt], bfr[kt], acc1);
    }
    int n = nt * 16 + ml;
    float bb = b2[n];
#pragma unroll
    for (int r = 0; r < 8; r++) {
      float m0v = acc0[r] + bb, m1v = acc1[r] + bb;
      unsafeAtomicAdd(&agg[(size_t)idx8[0][r] * 128 + n], m0v);
      unsafeAtomicAdd(&agg[(size_t)idx8[1][r] * 128 + n], m1v);
      stag[wave][(r + 8 * hi) * 128 + n]      = (half_t)m0v;
      stag[wave][(16 + r + 8 * hi) * 128 + n] = (half_t)m1v;
    }
  }

  // ---- GEMM3: coord hidden = silu(m @ Wc1 + cb1) ----
  v16h a3[2][4];
#pragma unroll
  for (int t = 0; t < 2; t++)
#pragma unroll
    for (int kt = 0; kt < 4; kt++) a3[t][kt] = load_afrag(&stag[wave][(t * 16 + ml) * 128], kt, hi);
#pragma unroll 2
  for (int nt = 0; nt < 8; nt++) {
    v16h bfr[4];
#pragma unroll
    for (int kt = 0; kt < 4; kt++) bfr[kt] = load_bfrag(Wc1, kt, nt, 8, lane);
    v8f acc0 = {}, acc1 = {};
#pragma unroll
    for (int kt = 0; kt < 4; kt++) {
      acc0 = wmma32(a3[0][kt], bfr[kt], acc0);
      acc1 = wmma32(a3[1][kt], bfr[kt], acc1);
    }
    int n = nt * 16 + ml;
    float bb = cb1[n];
#pragma unroll
    for (int r = 0; r < 8; r++) {
      stag[wave][(r + 8 * hi) * 128 + n]      = (half_t)siluf(acc0[r] + bb);
      stag[wave][(16 + r + 8 * hi) * 128 + n] = (half_t)siluf(acc1[r] + bb);
    }
  }

  // ---- coord scalar + position scatter (each lane: edge e0+lane) ----
  {
    float s = cb2[0];
    for (int n2 = 0; n2 < 128; n2++) s += (float)stag[wave][lane * 128 + n2] * wc2[n2];
    float cd = tanh_fast(s) * 0.1f;
    float inv = cd * fast_rcp(rn);
    unsafeAtomicAdd(&posDelta[rR * 3 + 0], dx * inv);
    unsafeAtomicAdd(&posDelta[rR * 3 + 1], dy * inv);
    unsafeAtomicAdd(&posDelta[rR * 3 + 2], dz * inv);
  }
}

// ---------------------------------------------------------------------------
// Generic fused GEMM: out = act( sum_b A_b @ W_b + bias ) (+resid)
// 2 M-tiles (32 rows) per wave; A_b: M x 128 f16, row stride sb halves
// (0 => broadcast row).  ACT: 0 none, 1 silu, 2 tanh.
// ---------------------------------------------------------------------------
template <int NB, int ACT, int NT2>
__global__ __launch_bounds__(256) void gemm_kernel(
    int M,
    const half_t* __restrict__ A0, int s0,
    const half_t* __restrict__ A1, int s1,
    const half_t* __restrict__ A2, int s2,
    const half_t* __restrict__ W0, const half_t* __restrict__ W1, const half_t* __restrict__ W2p,
    const float* __restrict__ bias, const float* __restrict__ resid,
    float* __restrict__ outF, half_t* __restrict__ outH) {
  int wave = threadIdx.x >> 5, lane = threadIdx.x & 31;
  int hi = lane >> 4, ml = lane & 15;
  int tile = blockIdx.x * 8 + wave;
  int m0 = tile * 32;
  if (m0 >= M) return;

  const half_t* As[3] = {A0, A1, A2};
  const int ss[3] = {s0, s1, s2};
  const half_t* Ws[3] = {W0, W1, W2p};

  v16h a[NB][2][4];
#pragma unroll
  for (int b = 0; b < NB; b++)
#pragma unroll
    for (int t = 0; t < 2; t++) {
      int mr = m0 + t * 16 + ml; if (mr >= M) mr = M - 1;
      const half_t* rb = As[b] + (size_t)mr * ss[b];
#pragma unroll
      for (int kt = 0; kt < 4; kt++) a[b][t][kt] = load_afrag(rb, kt, hi);
    }

  constexpr int N2 = NT2 * 16;
#pragma unroll 2
  for (int nt = 0; nt < NT2; nt++) {
    v16h bfr[NB * 4];
#pragma unroll
    for (int b = 0; b < NB; b++)
#pragma unroll
      for (int kt = 0; kt < 4; kt++) bfr[b * 4 + kt] = load_bfrag(Ws[b], kt, nt, NT2, lane);
    v8f acc0 = {}, acc1 = {};
#pragma unroll
    for (int b = 0; b < NB; b++)
#pragma unroll
      for (int kt = 0; kt < 4; kt++) {
        acc0 = wmma32(a[b][0][kt], bfr[b * 4 + kt], acc0);
        acc1 = wmma32(a[b][1][kt], bfr[b * 4 + kt], acc1);
      }
    int n = nt * 16 + ml;
    float bb = bias[n];
#pragma unroll
    for (int t = 0; t < 2; t++) {
      v8f& acc = (t == 0) ? acc0 : acc1;
#pragma unroll
      for (int r = 0; r < 8; r++) {
        int m = m0 + t * 16 + r + 8 * hi;
        if (m < M) {
          float v = acc[r] + bb;
          if (ACT == 1) v = siluf(v);
          else if (ACT == 2) v = tanh_fast(v);
          size_t off = (size_t)m * N2 + n;
          if (resid) v += resid[off];
          if (outF) outF[off] = v;
          if (outH) outH[off] = (half_t)v;
        }
      }
    }
  }
}

// ---------------------------------------------------------------------------
// Glue kernels (VALU; <1% of FLOPs)
// ---------------------------------------------------------------------------
__global__ void emb_kernel(const float* __restrict__ x, const float* __restrict__ W,
                           const float* __restrict__ b, float* __restrict__ hf32,
                           half_t* __restrict__ hf16, int n) {
  int i = blockIdx.x * blockDim.x + threadIdx.x;
  if (i >= n * 128) return;
  int r = i >> 7, c = i & 127;
  float s = b[c];
#pragma unroll
  for (int f = 0; f < 8; f++) s += x[(size_t)r * 8 + f] * W[f * 128 + c];
  hf32[i] = s; hf16[i] = (half_t)s;
}

__global__ void cvt_kernel(const float* __restrict__ x, half_t* __restrict__ y, int n) {
  int i = blockIdx.x * blockDim.x + threadIdx.x;
  if (i < n) y[i] = (half_t)x[i];
}

__global__ void posadd_kernel(float* __restrict__ p, const float* __restrict__ d, int n) {
  int i = blockIdx.x * blockDim.x + threadIdx.x;
  if (i < n) p[i] += d[i];
}

__global__ void colsum_kernel(const float* __restrict__ h, float* __restrict__ hsum, int n) {
  int c = threadIdx.x;
  int r0 = blockIdx.x * 400;
  int r1 = r0 + 400; if (r1 > n) r1 = n;
  float s = 0.f;
  for (int r = r0; r < r1; r++) s += h[(size_t)r * 128 + c];
  unsafeAtomicAdd(&hsum[c], s);
}

__global__ void global_mlp_kernel(const float* __restrict__ hsum, const float* __restrict__ W1,
                                  const float* __restrict__ b1, const float* __restrict__ W2,
                                  const float* __restrict__ b2, half_t* __restrict__ g16, float invN) {
  __shared__ float mean[128], hid[128];
  int c = threadIdx.x;
  mean[c] = hsum[c] * invN;
  __syncthreads();
  float s = b1[c];
  for (int k = 0; k < 128; k++) s += mean[k] * W1[k * 128 + c];
  hid[c] = siluf(s);
  __syncthreads();
  float o = b2[c];
  for (int k = 0; k < 128; k++) o += hid[k] * W2[k * 128 + c];
  g16[c] = (half_t)o;
}

__global__ void posmlp_kernel(const float* __restrict__ pos, const float* __restrict__ W1,
                              const float* __restrict__ b1, const float* __restrict__ W2,
                              const float* __restrict__ b2, half_t* __restrict__ pf16, int n) {
  int nd = blockIdx.x * blockDim.x + threadIdx.x;
  if (nd >= n) return;
  float p0 = pos[nd * 3 + 0], p1 = pos[nd * 3 + 1], p2 = pos[nd * 3 + 2];
  float h1[32];
#pragma unroll
  for (int k = 0; k < 32; k++)
    h1[k] = siluf(b1[k] + p0 * W1[k] + p1 * W1[32 + k] + p2 * W1[64 + k]);
  for (int c = 0; c < 128; c++) {
    float s = b2[c];
#pragma unroll
    for (int k = 0; k < 32; k++) s += h1[k] * W2[k * 128 + c];
    pf16[(size_t)nd * 128 + c] = (half_t)s;
  }
}

__global__ void ln_kernel(float* __restrict__ h, const float* __restrict__ fused,
                          const float* __restrict__ g, const float* __restrict__ b,
                          half_t* __restrict__ hf16, int n) {
  int nd = blockIdx.x * blockDim.x + threadIdx.x;
  if (nd >= n) return;
  size_t o = (size_t)nd * 128;
  float mu = 0.f;
  for (int c = 0; c < 128; c++) mu += h[o + c] + fused[o + c];
  mu *= (1.f / 128.f);
  float var = 0.f;
  for (int c = 0; c < 128; c++) { float x = h[o + c] + fused[o + c] - mu; var += x * x; }
  var *= (1.f / 128.f);
  float is = rsqrtf(var + 1e-5f);
  for (int c = 0; c < 128; c++) {
    float y = (h[o + c] + fused[o + c] - mu) * is * g[c] + b[c];
    h[o + c] = y; hf16[o + c] = (half_t)y;
  }
}

__global__ void rowdot_kernel(const half_t* __restrict__ t, const float* __restrict__ w,
                              const float* __restrict__ b, float* __restrict__ scores, int n) {
  int nd = blockIdx.x * blockDim.x + threadIdx.x;
  if (nd >= n) return;
  float s = b[0];
  for (int c = 0; c < 128; c++) s += (float)t[(size_t)nd * 128 + c] * w[c];
  scores[nd] = s;
}

__global__ __launch_bounds__(1024) void softpool_kernel(const float* __restrict__ scores,
                                                        const float* __restrict__ h,
                                                        float* __restrict__ pooled, int n) {
  __shared__ float red[1024];
  int t = threadIdx.x;
  float mx = -3.4e38f;
  for (int i = t; i < n; i += 1024) mx = fmaxf(mx, scores[i]);
  red[t] = mx; __syncthreads();
  for (int s2 = 512; s2 > 0; s2 >>= 1) { if (t < s2) red[t] = fmaxf(red[t], red[t + s2]); __syncthreads(); }
  float gmax = red[0]; __syncthreads();
  float sum = 0.f;
  for (int i = t; i < n; i += 1024) sum += __expf(scores[i] - gmax);
  red[t] = sum; __syncthreads();
  for (int s2 = 512; s2 > 0; s2 >>= 1) { if (t < s2) red[t] += red[t + s2]; __syncthreads(); }
  float Z = red[0]; __syncthreads();
  int c = t & 127, gidx = t >> 7;
  float acc = 0.f;
  for (int i = gidx; i < n; i += 8) acc += __expf(scores[i] - gmax) * h[(size_t)i * 128 + c];
  red[t] = acc; __syncthreads();
  if (gidx == 0) {
    float s = 0.f;
    for (int g2 = 0; g2 < 8; g2++) s += red[g2 * 128 + c];
    pooled[c] = s / Z;
  }
}

__global__ __launch_bounds__(256) void outmlp_kernel(const float* __restrict__ pooled,
                                                     const float* __restrict__ W1, const float* __restrict__ b1,
                                                     const float* __restrict__ W2, const float* __restrict__ b2,
                                                     float* __restrict__ out) {
  __shared__ float hid[128];
  int t = threadIdx.x;
  if (t < 128) {
    float s = b1[t];
    for (int k = 0; k < 128; k++) s += pooled[k] * W1[k * 128 + t];
    hid[t] = siluf(s);
  }
  __syncthreads();
  float o = b2[t];
  for (int k = 0; k < 128; k++) o += hid[k] * W2[k * 256 + t];
  out[t] = o;
}

// ---------------------------------------------------------------------------
// Host driver
// ---------------------------------------------------------------------------
extern "C" void kernel_launch(void* const* d_in, const int* in_sizes, int n_in,
                              void* d_out, int out_size, void* d_ws, size_t ws_size,
                              hipStream_t stream) {
  (void)n_in; (void)out_size; (void)ws_size;
  // Canonical slots (== reference insertion order):
  // 0 x, 1 pos, 2 edge_index, 3-4 emb(W,b),
  // layer l base 5+12l: eW1,eB1,eW2,eB2, nW1,nB1,nW2,nB2, cW1,cB1,cW2,cB2
  // 41-44 local, 45-48 global, 49-52 pos_mlp, 53-56 fusion, 57-60 attn, 61-64 out, 65-66 ln
  int map[67];
  bool sortedOrder = (in_sizes[0] == 2 * Ee);  // JAX pytree (sorted keys) puts edge_index first
  if (!sortedOrder) {
    for (int i = 0; i < 67; i++) map[i] = i;
  } else {
    map[0] = 66; map[1] = 65; map[2] = 0; map[3] = 5; map[4] = 6;
    for (int l = 0; l < 3; l++) {
      int cb = 5 + 12 * l, ib = 15 + 12 * l;  // sorted layer dict: coord, edge, node
      for (int i = 0; i < 4; i++) map[cb + i] = ib + 4 + i;       // edge
      for (int i = 0; i < 4; i++) map[cb + 4 + i] = ib + 8 + i;   // node
      for (int i = 0; i < 4; i++) map[cb + 8 + i] = ib + i;       // coord
    }
    for (int i = 0; i < 4; i++) map[41 + i] = 53 + i;  // local
    for (int i = 0; i < 4; i++) map[45 + i] = 11 + i;  // global
    for (int i = 0; i < 4; i++) map[49 + i] = 61 + i;  // pos_mlp
    for (int i = 0; i < 4; i++) map[53 + i] = 7 + i;   // fusion
    for (int i = 0; i < 4; i++) map[57 + i] = 1 + i;   // attn
    for (int i = 0; i < 4; i++) map[61 + i] = 57 + i;  // out
    map[65] = 51; map[66] = 52;
  }
  auto FP = [&](int c) -> const float* { return (const float*)d_in[map[c]]; };
  const int* edge_index = (const int*)d_in[map[2]];

  // ---- workspace carve ----
  uintptr_t base = (uintptr_t)d_ws;
  auto alloc = [&](size_t bytes) -> void* {
    uintptr_t p = (base + 255) & ~(uintptr_t)255;
    base = p + bytes;
    return (void*)p;
  };
  float*  hf32    = (float*)alloc((size_t)Nn * 128 * 4);
  half_t* hf16    = (half_t*)alloc((size_t)Nn * 128 * 2);
  float*  aggf32  = (float*)alloc((size_t)Nn * 128 * 4);   // also reused as "fused"
  half_t* aggf16  = (half_t*)alloc((size_t)Nn * 128 * 2);
  half_t* tmp16a  = (half_t*)alloc((size_t)Nn * 128 * 2);
  half_t* tmp16b  = (half_t*)alloc((size_t)Nn * 128 * 2);
  half_t* pf16    = (half_t*)alloc((size_t)Nn * 128 * 2);
  float*  posCur  = (float*)alloc((size_t)Nn * 3 * 4);
  float*  posDel  = (float*)alloc((size_t)Nn * 3 * 4);
  float*  scores  = (float*)alloc((size_t)Nn * 4);
  float*  hsum    = (float*)alloc(128 * 4);
  half_t* g16     = (half_t*)alloc(128 * 2);
  float*  pooled  = (float*)alloc(128 * 4);

  auto packbuf = [&]() -> half_t* { return (half_t*)alloc(128 * 128 * 2); };
  auto pack = [&](const float* W, half_t* dst) {
    pack_kernel<<<(128 * 128 + 255) / 256, 256, 0, stream>>>(W, dst, 128, 128);
  };

  half_t *pe1a[3], *pe1b[3], *pe2[3], *pc1[3], *pn1a[3], *pn1b[3], *pn2[3];
  for (int l = 0; l < 3; l++) {
    int cb = 5 + 12 * l;
    pe1a[l] = packbuf(); pack(FP(cb + 0),             pe1a[l]);   // edge W1 rows   0..127
    pe1b[l] = packbuf(); pack(FP(cb + 0) + 128 * 128, pe1b[l]);   // edge W1 rows 128..255
    pe2[l]  = packbuf(); pack(FP(cb + 2),             pe2[l]);
    pc1[l]  = packbuf(); pack(FP(cb + 8),             pc1[l]);
    pn1a[l] = packbuf(); pack(FP(cb + 4),             pn1a[l]);   // node W1 rows   0..127 (h)
    pn1b[l] = packbuf(); pack(FP(cb + 4) + 128 * 128, pn1b[l]);   // node W1 rows 128..255 (agg)
    pn2[l]  = packbuf(); pack(FP(cb + 6),             pn2[l]);
  }
  half_t* pl1 = packbuf(); pack(FP(41), pl1);
  half_t* pl2 = packbuf(); pack(FP(43), pl2);
  half_t* pf1a = packbuf(); pack(FP(53),                 pf1a);   // fusion W1 rows   0..127 (local)
  half_t* pf1b = packbuf(); pack(FP(53) + 128 * 128,     pf1b);   //               128..255 (global)
  half_t* pf1c = packbuf(); pack(FP(53) + 2 * 128 * 128, pf1c);   //               256..383 (pos feat)
  half_t* pf2  = packbuf(); pack(FP(55), pf2);
  half_t* pa1  = packbuf(); pack(FP(57), pa1);

  // ---- embedding + initial positions ----
  emb_kernel<<<(Nn * 128 + 255) / 256, 256, 0, stream>>>(FP(0), FP(3), FP(4), hf32, hf16, Nn);
  hipMemcpyAsync(posCur, FP(1), (size_t)Nn * 3 * 4, hipMemcpyDeviceToDevice, stream);

  const int tiles32 = (Nn + 31) / 32;
  const int gridG = (tiles32 + 7) / 8;

  // ---- EGNN layers ----
  for (int l = 0; l < 3; l++) {
    int cb = 5 + 12 * l;
    hipMemsetAsync(aggf32, 0, (size_t)Nn * 128 * 4, stream);
    hipMemsetAsync(posDel, 0, (size_t)Nn * 3 * 4, stream);
    edge_kernel<<<Ee / 128, 128, 0, stream>>>(
        hf16, posCur, edge_index, edge_index + Ee,
        pe1a[l], pe1b[l], FP(cb + 0) + 256 * 128, FP(cb + 1),
        pe2[l], FP(cb + 3),
        pc1[l], FP(cb + 9), FP(cb + 10), FP(cb + 11),
        aggf32, posDel, Ee);
    cvt_kernel<<<(Nn * 128 + 255) / 256, 256, 0, stream>>>(aggf32, aggf16, Nn * 128);
    // node MLP: t = silu([h,agg]@W1+b1); h += t@W2+b2
    gemm_kernel<2, 1, 8><<<gridG, 256, 0, stream>>>(
        Nn, hf16, 128, aggf16, 128, (const half_t*)nullptr, 0,
        pn1a[l], pn1b[l], (const half_t*)nullptr, FP(cb + 5),
        (const float*)nullptr, (float*)nullptr, tmp16b);
    gemm_kernel<1, 0, 8><<<gridG, 256, 0, stream>>>(
        Nn, tmp16b, 128, (const half_t*)nullptr, 0, (const half_t*)nullptr, 0,
        pn2[l], (const half_t*)nullptr, (const half_t*)nullptr, FP(cb + 7),
        hf32, hf32, hf16);
    posadd_kernel<<<(Nn * 3 + 255) / 256, 256, 0, stream>>>(posCur, posDel, Nn * 3);
  }

  // ---- readout ----
  // local = mlp(h)
  gemm_kernel<1, 1, 8><<<gridG, 256, 0, stream>>>(
      Nn, hf16, 128, (const half_t*)nullptr, 0, (const half_t*)nullptr, 0,
      pl1, (const half_t*)nullptr, (const half_t*)nullptr, FP(42),
      (const float*)nullptr, (float*)nullptr, tmp16a);
  gemm_kernel<1, 0, 8><<<gridG, 256, 0, stream>>>(
      Nn, tmp16a, 128, (const half_t*)nullptr, 0, (const half_t*)nullptr, 0,
      pl2, (const half_t*)nullptr, (const half_t*)nullptr, FP(44),
      (const float*)nullptr, (float*)nullptr, tmp16b);
  // pos features
  posmlp_kernel<<<(Nn + 255) / 256, 256, 0, stream>>>(posCur, FP(49), FP(50), FP(51), FP(52), pf16, Nn);
  // global: mean(h) -> mlp -> broadcast vector
  hipMemsetAsync(hsum, 0, 128 * 4, stream);
  colsum_kernel<<<(Nn + 399) / 400, 128, 0, stream>>>(hf32, hsum, Nn);
  global_mlp_kernel<<<1, 128, 0, stream>>>(hsum, FP(45), FP(46), FP(47), FP(48), g16, 1.0f / Nn);
  // fusion
  gemm_kernel<3, 1, 8><<<gridG, 256, 0, stream>>>(
      Nn, tmp16b, 128, g16, 0, pf16, 128,
      pf1a, pf1b, pf1c, FP(54),
      (const float*)nullptr, (float*)nullptr, tmp16a);
  gemm_kernel<1, 0, 8><<<gridG, 256, 0, stream>>>(
      Nn, tmp16a, 128, (const half_t*)nullptr, 0, (const half_t*)nullptr, 0,
      pf2, (const half_t*)nullptr, (const half_t*)nullptr, FP(56),
      (const float*)nullptr, aggf32, (half_t*)nullptr);
  // h = LN(h + fused)
  ln_kernel<<<(Nn + 255) / 256, 256, 0, stream>>>(hf32, aggf32, FP(65), FP(66), hf16, Nn);
  // attention pooling
  gemm_kernel<1, 2, 8><<<gridG, 256, 0, stream>>>(
      Nn, hf16, 128, (const half_t*)nullptr, 0, (const half_t*)nullptr, 0,
      pa1, (const half_t*)nullptr, (const half_t*)nullptr, FP(58),
      (const float*)nullptr, (float*)nullptr, tmp16a);
  rowdot_kernel<<<(Nn + 255) / 256, 256, 0, stream>>>(tmp16a, FP(59), FP(60), scores, Nn);
  softpool_kernel<<<1, 1024, 0, stream>>>(scores, hf32, pooled, Nn);
  // output MLP (M=1, VALU)
  outmlp_kernel<<<1, 256, 0, stream>>>(pooled, FP(61), FP(62), FP(63), FP(64), (float*)d_out);
}
